// MultiHeadAttention_64458869178973
// MI455X (gfx1250) — matrix operand order repaired and checked by reading, not compiled, not run on hardware
//
#include <hip/hip_runtime.h>
#include <hip/hip_bf16.h>
#include <cstdint>

// ---------------------------------------------------------------------------
// MI455X (gfx1250) multi-head attention forward, f16 WMMA path + TDM staging.
// B=4, T=2048, C=2048, H=16, D=128.
// ---------------------------------------------------------------------------

typedef _Float16 v16h __attribute__((ext_vector_type(16)));
typedef _Float16 v8h  __attribute__((ext_vector_type(8)));
typedef _Float16 v4h  __attribute__((ext_vector_type(4)));
typedef float    v8f  __attribute__((ext_vector_type(8)));
typedef float    v4f  __attribute__((ext_vector_type(4)));
typedef unsigned int u32x4 __attribute__((ext_vector_type(4)));
typedef int          i32x4 __attribute__((ext_vector_type(4)));
typedef int          i32x8 __attribute__((ext_vector_type(8)));

static constexpr int NB = 4;      // batch
static constexpr int NT = 2048;   // sequence
static constexpr int NC = 2048;   // channels
static constexpr int NH = 16;     // heads
static constexpr int ND = 128;    // head dim
static constexpr int MROWS = NB * NT;  // 8192

__device__ __forceinline__ v8f wmma_f16(v16h a, v16h b, v8f c) {
  // D = A(16x32 f16) * B(32x16 f16) + C(16x16 f32)
  return __builtin_amdgcn_wmma_f32_16x16x32_f16(
      /*neg_a=*/false, a, /*neg_b=*/false, b,
      /*c_mod=*/(short)0, c, /*reuse_a=*/false, /*reuse_b=*/false);
}

__device__ __forceinline__ v16h cat8(v8h lo, v8h hi) {
  return __builtin_shufflevector(lo, hi, 0, 1, 2, 3, 4, 5, 6, 7,
                                          8, 9, 10, 11, 12, 13, 14, 15);
}

// ---------------------------------------------------------------------------
// Tensor Data Mover: DMA a (tile_d1 x tile_d0) f16 tile (row stride stride0
// elements) from global into LDS at ldsAddr (contiguous row-major).
// D# packing per CDNA5 ISA 8.3/8.4 (group0: count/lds/global/type,
// group1: data_size=2B, dims, tile dims, stride). Groups 2/3 zero (2D tile).
// ---------------------------------------------------------------------------
__device__ __forceinline__ void tdm_load_tile_f16(const _Float16* gsrc,
                                                  unsigned ldsAddr,
                                                  int tensor_d0, int tensor_d1,
                                                  int tile_d0, int tile_d1,
                                                  int stride0) {
  const unsigned long long ga = (unsigned long long)(uintptr_t)gsrc;
  u32x4 g0;
  g0[0] = 1u;                                   // count=1, user descriptor
  g0[1] = ldsAddr;                              // lds byte address
  g0[2] = (unsigned)(ga & 0xffffffffull);       // global_addr[31:0]
  g0[3] = (unsigned)((ga >> 32) & 0x1ffffffull) // global_addr[56:32]
          | (2u << 30);                         // type=2 ("image")
  i32x8 g1;
  g1[0] = (int)(1u << 16);                      // data_size=1 -> 2 bytes
  g1[1] = (int)((unsigned)tensor_d0 << 16);     // tensor_dim0[15:0]
  g1[2] = (int)(((unsigned)tensor_d0 >> 16) | ((unsigned)tensor_d1 << 16));
  g1[3] = (int)((((unsigned)tensor_d1 >> 16) & 0xffffu) |
                ((unsigned)tile_d0 << 16));     // tile_dim0
  g1[4] = (int)((unsigned)tile_d1 & 0xffffu);   // tile_dim1 (tile_dim2=0)
  g1[5] = stride0;                              // tensor_dim0_stride[31:0]
  g1[6] = 0;
  g1[7] = 0;
  const i32x4 z4 = {0, 0, 0, 0};
#if __clang_major__ >= 23
  const i32x8 z8 = {0, 0, 0, 0, 0, 0, 0, 0};
  __builtin_amdgcn_tensor_load_to_lds(g0, g1, z4, z4, z8, 0);
#else
  __builtin_amdgcn_tensor_load_to_lds(g0, g1, z4, z4, 0);
#endif
}

// ---------------------------------------------------------------------------
// Tiled WMMA GEMM:  C[M,N] = A[M,K] * B[K,N] + bias[N]
// Block tile 128(M) x 128(N), K-step 32, 256 threads = 8 waves in a 4(M)x2(N)
// grid; each wave owns a 32x64 tile = 2x4 accumulators, 8 WMMAs per K-step.
// A fp32 (VGPR-convert staging) or f16 (TDM async staging); B fp32 converted
// and stored transposed [n][k] in LDS.
// ---------------------------------------------------------------------------
template <bool A_HALF, bool STORE_HALF>
__global__ __launch_bounds__(256) void gemm_bias_wmma(
    const void* __restrict__ Ap, const float* __restrict__ Bp,
    const float* __restrict__ bias, void* __restrict__ Cp,
    int M, int N, int K) {
  __shared__ _Float16 As[128][32];   // row-major M x K tile
  __shared__ _Float16 Bs[128][32];   // transposed: [n][k]

  const int tid  = threadIdx.x;
  const int wave = tid >> 5;
  const int lane = tid & 31;
  const int ln   = lane & 15;
  const int g    = lane >> 4;
  const int mw   = wave >> 1;        // 0..3 (M group)
  const int nw   = wave & 1;         // 0..1 (N group)

  const int m0 = blockIdx.y * 128;
  const int n0 = blockIdx.x * 128;

  const v8f vzero = {0.f, 0.f, 0.f, 0.f, 0.f, 0.f, 0.f, 0.f};
  v8f acc[2][4];
#pragma unroll
  for (int i = 0; i < 2; ++i)
#pragma unroll
    for (int j = 0; j < 4; ++j) acc[i][j] = vzero;

  const unsigned ldsAs = (unsigned)(uintptr_t)&As[0][0];

  for (int k0 = 0; k0 < K; k0 += 32) {
    // ---- stage A tile (128x32) ----
    if (A_HALF) {
      if (wave == 0) {
        tdm_load_tile_f16((const _Float16*)Ap + (size_t)m0 * K + k0, ldsAs,
                          /*tensor_d0=*/K, /*tensor_d1=*/M,
                          /*tile_d0=*/32, /*tile_d1=*/128, /*stride0=*/K);
        __builtin_amdgcn_s_wait_tensorcnt(0);
      }
    } else {
#pragma unroll
      for (int p = 0; p < 4; ++p) {
        const int r = p * 32 + (tid >> 3);
        const int c = (tid & 7) * 4;
        const float* A = (const float*)Ap;
        v4f d = *(const v4f*)(A + (size_t)(m0 + r) * K + k0 + c);
        v4h h;
#pragma unroll
        for (int i = 0; i < 4; ++i) h[i] = (_Float16)d[i];
        *(v4h*)&As[r][c] = h;
      }
    }
    // ---- stage B tile (32x128), store transposed [n][k] ----
#pragma unroll
    for (int p = 0; p < 4; ++p) {
      const int kr = p * 8 + (tid >> 5);
      const int nc = (tid & 31) * 4;
      v4f d = *(const v4f*)(Bp + (size_t)(k0 + kr) * N + n0 + nc);
#pragma unroll
      for (int i = 0; i < 4; ++i) Bs[nc + i][kr] = (_Float16)d[i];
    }
    __syncthreads();

    // ---- compute: 8 WMMAs per wave per K-step ----
    v16h a[2];
#pragma unroll
    for (int i = 0; i < 2; ++i) {
      const int row = mw * 32 + i * 16 + ln;
      v8h alo = *(const v8h*)&As[row][g * 16];
      v8h ahi = *(const v8h*)&As[row][g * 16 + 8];
      a[i] = cat8(alo, ahi);
    }
#pragma unroll
    for (int nt = 0; nt < 4; ++nt) {
      v16h b = *(const v16h*)&Bs[nw * 64 + nt * 16 + ln][g * 16];
      acc[0][nt] = wmma_f16(a[0], b, acc[0][nt]);
      acc[1][nt] = wmma_f16(a[1], b, acc[1][nt]);
    }
    __syncthreads();
  }

  // ---- epilogue: C layout is (M = r + 8*g, N = ln) per VGPR r ----
#pragma unroll
  for (int nt = 0; nt < 4; ++nt) {
    const int col = n0 + nw * 64 + nt * 16 + ln;
    const float bv = bias[col];
#pragma unroll
    for (int i = 0; i < 2; ++i) {
#pragma unroll
      for (int r = 0; r < 8; ++r) {
        const int row = m0 + mw * 32 + i * 16 + r + 8 * g;
        const float v = acc[i][nt][r] + bv;
        if (STORE_HALF)
          ((_Float16*)Cp)[(size_t)row * N + col] = (_Float16)v;
        else
          ((float*)Cp)[(size_t)row * N + col] = v;
      }
    }
  }
}

// ---------------------------------------------------------------------------
// RoPE + layout shuffle.
// qkv (B,T,3,H,D) f16  ->  Q (B,H,T,D) f16 (scaled by 1/sqrt(D), rotated),
//                          K (B,H,T,D) f16 (rotated),
//                          VT (B,H,D,T) f16 (transposed copy).
// One block per (b,h,t), 64 threads = one rotation pair (d, d+64) each.
// ---------------------------------------------------------------------------
__global__ __launch_bounds__(64) void rope_reorder(
    const _Float16* __restrict__ qkv, _Float16* __restrict__ Q,
    _Float16* __restrict__ Kd, _Float16* __restrict__ VT) {
  const int d = threadIdx.x;                 // 0..63
  int idx = blockIdx.x;
  const int t = idx & (NT - 1);
  idx >>= 11;                                // NT = 2048
  const int h = idx & (NH - 1);
  const int b = idx >> 4;                    // NH = 16

  const size_t base = ((size_t)(b * NT + t)) * (3 * NC);
  const _Float16* qp = qkv + base + 0 * NC + h * ND;
  const _Float16* kp = qkv + base + 1 * NC + h * ND;
  const _Float16* vp = qkv + base + 2 * NC + h * ND;

  const float q1 = (float)qp[d], q2 = (float)qp[d + 64];
  const float k1 = (float)kp[d], k2 = (float)kp[d + 64];

  // theta = base^(-2d/D) = exp(-ln(10000) * d / 64)
  const float theta = __expf(-9.210340371976184f * ((float)d * (1.0f / 64.0f)));
  const float ang = (float)t * theta;
  const float s = sinf(ang);
  const float c = cosf(ang);
  const float scale = 0.08838834764831845f;  // 1/sqrt(128), folded into Q

  const size_t qkb = ((size_t)((b * NH + h)) * NT + t) * ND;
  Q[qkb + d]       = (_Float16)((q1 * c - q2 * s) * scale);
  Q[qkb + d + 64]  = (_Float16)((q1 * s + q2 * c) * scale);
  Kd[qkb + d]      = (_Float16)(k1 * c - k2 * s);
  Kd[qkb + d + 64] = (_Float16)(k1 * s + k2 * c);

  const size_t vtb = (size_t)(b * NH + h) * ND * NT;
  VT[vtb + (size_t)d * NT + t]        = vp[d];
  VT[vtb + (size_t)(d + 64) * NT + t] = vp[d + 64];
}

// ---------------------------------------------------------------------------
// Flash attention, causal. One wave (32 threads) per (b,h, 16-query tile).
// Per 32-key block: 8 WMMAs for S = Q*K^T, fp32 online softmax (xor-shuffle
// row reductions within 16-lane halves), P relaid via LDS into A-fragment
// layout, 8 WMMAs for O += P*V. Output (B,T,C) f16.
// ---------------------------------------------------------------------------
__global__ __launch_bounds__(32) void attn_flash_wmma(
    const _Float16* __restrict__ Q, const _Float16* __restrict__ Kd,
    const _Float16* __restrict__ VT, _Float16* __restrict__ Oh) {
  __shared__ _Float16 Pl[16][32];

  const int lane = threadIdx.x;
  const int ln = lane & 15;
  const int g  = lane >> 4;
  const int qblk = blockIdx.x & 127;          // NT/16 = 128
  const int bh   = blockIdx.x >> 7;
  const int b = bh >> 4, h = bh & 15;

  // Q A-fragments for 4 d-chunks of 32 (loaded once, pre-scaled by 1/sqrt(D))
  v16h qa[4];
  {
    const _Float16* qr = Q + ((size_t)bh * NT + qblk * 16 + ln) * ND;
#pragma unroll
    for (int c = 0; c < 4; ++c) {
      v8h lo = *(const v8h*)(qr + c * 32 + g * 16);
      v8h hi = *(const v8h*)(qr + c * 32 + g * 16 + 8);
      qa[c] = cat8(lo, hi);
    }
  }

  const v8f vzero = {0.f, 0.f, 0.f, 0.f, 0.f, 0.f, 0.f, 0.f};
  v8f O[8];
#pragma unroll
  for (int dt = 0; dt < 8; ++dt) O[dt] = vzero;
  float mrow[8], lrow[8];
#pragma unroll
  for (int r = 0; r < 8; ++r) { mrow[r] = -1e30f; lrow[r] = 0.f; }

  const int kend = (qblk + 1) * 16;           // causal upper bound (exclusive)
  for (int kb = 0; kb * 32 < kend; ++kb) {
    const int kbase = kb * 32;

    // ---- S = Q * K^T  (two 16x16 tiles over a 32-key block) ----
    v8f S0 = vzero, S1 = vzero;
    const _Float16* kr0 = Kd + ((size_t)bh * NT + kbase + ln) * ND;
    const _Float16* kr1 = kr0 + (size_t)16 * ND;
#pragma unroll
    for (int c = 0; c < 4; ++c) {
      v16h b0 = *(const v16h*)(kr0 + c * 32 + g * 16);
      v16h b1 = *(const v16h*)(kr1 + c * 32 + g * 16);
      S0 = wmma_f16(qa[c], b0, S0);
      S1 = wmma_f16(qa[c], b1, S1);
    }

    // ---- causal mask + online softmax (fp32) ----
    float p0[8], p1[8];
#pragma unroll
    for (int r = 0; r < 8; ++r) {
      const int qg = qblk * 16 + r + 8 * g;
      p0[r] = (kbase + ln      <= qg) ? S0[r] : -1e30f;
      p1[r] = (kbase + 16 + ln <= qg) ? S1[r] : -1e30f;
    }
#pragma unroll
    for (int r = 0; r < 8; ++r) {
      float mx = fmaxf(p0[r], p1[r]);
#pragma unroll
      for (int off = 8; off >= 1; off >>= 1)
        mx = fmaxf(mx, __shfl_xor(mx, off, 32));
      const float mn = fmaxf(mrow[r], mx);
      const float fr = __expf(mrow[r] - mn);
      mrow[r] = mn;
      const float e0 = __expf(p0[r] - mn);
      const float e1 = __expf(p1[r] - mn);
      p0[r] = e0;
      p1[r] = e1;
      float rs = e0 + e1;
#pragma unroll
      for (int off = 8; off >= 1; off >>= 1)
        rs += __shfl_xor(rs, off, 32);
      lrow[r] = lrow[r] * fr + rs;
#pragma unroll
      for (int dt = 0; dt < 8; ++dt) O[dt][r] = O[dt][r] * fr;
    }

    // ---- P: accumulator layout -> A-fragment layout via LDS ----
#pragma unroll
    for (int r = 0; r < 8; ++r) {
      Pl[r + 8 * g][ln]      = (_Float16)p0[r];
      Pl[r + 8 * g][16 + ln] = (_Float16)p1[r];
    }
    __syncthreads();  // single-wave block: LDS wait (barrier itself = NOP)
    v8h plo = *(const v8h*)&Pl[ln][g * 16];
    v8h phi = *(const v8h*)&Pl[ln][g * 16 + 8];
    v16h pa = cat8(plo, phi);
    __syncthreads();

    // ---- O += P * V  (V pre-transposed: contiguous keys per d-row) ----
#pragma unroll
    for (int dt = 0; dt < 8; ++dt) {
      v16h vf =
          *(const v16h*)(VT + ((size_t)bh * ND + dt * 16 + ln) * NT + kbase + g * 16);
      O[dt] = wmma_f16(pa, vf, O[dt]);
    }
  }

  // ---- normalize + store (B,T,C) f16 ----
#pragma unroll
  for (int dt = 0; dt < 8; ++dt) {
    const int col = h * ND + dt * 16 + ln;
#pragma unroll
    for (int r = 0; r < 8; ++r) {
      const int trow = qblk * 16 + r + 8 * g;
      Oh[((size_t)b * NT + trow) * NC + col] = (_Float16)(O[dt][r] / lrow[r]);
    }
  }
}

// ---------------------------------------------------------------------------
// Host-side orchestration.
// Workspace layout (halves):
//   qkv_h : MROWS * 3*NC   (x @ Wqkv + bqkv, f16)
//   Qh    : NB*NH*NT*ND    (rope'd, scaled)
//   Kh    : NB*NH*NT*ND
//   VTh   : NB*NH*ND*NT    (transposed V)
//   AOh   : MROWS * NC     (attention output, f16)
// ---------------------------------------------------------------------------
extern "C" void kernel_launch(void* const* d_in, const int* in_sizes, int n_in,
                              void* d_out, int out_size, void* d_ws,
                              size_t ws_size, hipStream_t stream) {
  const float* x    = (const float*)d_in[0];
  const float* Wqkv = (const float*)d_in[1];
  const float* bqkv = (const float*)d_in[2];
  const float* Wo   = (const float*)d_in[3];
  const float* bo   = (const float*)d_in[4];
  float* out = (float*)d_out;

  _Float16* qkv_h = (_Float16*)d_ws;
  _Float16* Qh  = qkv_h + (size_t)MROWS * 3 * NC;
  _Float16* Kh  = Qh  + (size_t)NB * NH * NT * ND;
  _Float16* VTh = Kh  + (size_t)NB * NH * NT * ND;
  _Float16* AOh = VTh + (size_t)NB * NH * ND * NT;

  // 1) QKV projection: (8192 x 2048) * (2048 x 6144) + bias -> f16
  gemm_bias_wmma<false, true>
      <<<dim3((3 * NC) / 128, MROWS / 128), 256, 0, stream>>>(
          x, Wqkv, bqkv, qkv_h, MROWS, 3 * NC, NC);

  // 2) RoPE + (B,H,T,D)/(B,H,D,T) relayout
  rope_reorder<<<NB * NH * NT, 64, 0, stream>>>(qkv_h, Qh, Kh, VTh);

  // 3) Causal flash attention
  attn_flash_wmma<<<NB * NH * (NT / 16), 32, 0, stream>>>(Qh, Kh, VTh, AOh);

  // 4) Output projection: (8192 x 2048) * (2048 x 2048) + bias -> fp32
  //    A operand staged by the Tensor Data Mover (async DMA into LDS).
  gemm_bias_wmma<true, false>
      <<<dim3(NC / 128, MROWS / 128), 256, 0, stream>>>(
          AOh, Wo, bo, out, MROWS, NC, NC);
}
